// MinibatchDiscrimination_42073499631717
// MI455X (gfx1250) — matrix operand order repaired and checked by reading, not compiled, not run on hardware
//
#include <hip/hip_runtime.h>
#include <hip/hip_bf16.h>
#include <math.h>

// ---------------------------------------------------------------------------
// MinibatchDiscrimination for MI455X (gfx1250, wave32, WMMA)
//   x : [256, 1024] f32
//   T : [1024, 3200] f32   (3200 = 64 out_features * 50 kernel_dims)
//   out: [256, 1088] f32 = concat(x, sum_j exp(-L1(M_i, M_j)) - 1)
// ---------------------------------------------------------------------------

#define B_SZ   256
#define IN_F   1024
#define OUT_F  64
#define KD     50
#define NCOLS  (OUT_F * KD)      // 3200
#define OUT_W  (IN_F + OUT_F)    // 1088

typedef _Float16 v8h  __attribute__((ext_vector_type(8)));
typedef _Float16 v16h __attribute__((ext_vector_type(16)));
typedef float    v8f  __attribute__((ext_vector_type(8)));

// ---------------------------------------------------------------------------
// Kernel 1: convert x -> f16 (A operand) and copy x into out[:, 0:1024]
// ---------------------------------------------------------------------------
__global__ void convert_x_kernel(const float* __restrict__ x,
                                 _Float16* __restrict__ a16,
                                 float* __restrict__ out)
{
    int idx = blockIdx.x * blockDim.x + threadIdx.x;   // 0 .. 256*1024-1
    if (idx >= B_SZ * IN_F) return;
    float v = x[idx];
    a16[idx] = (_Float16)v;
    int row = idx >> 10;          // / 1024
    int col = idx & (IN_F - 1);   // % 1024
    out[row * OUT_W + col] = v;
}

// ---------------------------------------------------------------------------
// Kernel 2: LDS-tiled transpose + convert: T[1024,3200] f32 -> Bt[3200,1024] f16
//   Bt[n][k] = T[k][n].  Row-major-in-K Bt makes WMMA B-fragment loads
//   contiguous per lane.
// ---------------------------------------------------------------------------
__global__ void transpose_T_kernel(const float* __restrict__ T,
                                   _Float16* __restrict__ bt)
{
    __shared__ float tile[32][33];          // +1 pad kills bank conflicts
    const int tx = threadIdx.x;             // 0..31
    const int ty = threadIdx.y;             // 0..7
    const int n0 = blockIdx.x * 32;         // N tile base (3200/32 = 100)
    const int k0 = blockIdx.y * 32;         // K tile base (1024/32 = 32)

    #pragma unroll
    for (int r = 0; r < 4; ++r) {
        int k = k0 + ty + r * 8;
        tile[ty + r * 8][tx] = T[k * NCOLS + (n0 + tx)];   // coalesced read
    }
    __syncthreads();
    #pragma unroll
    for (int r = 0; r < 4; ++r) {
        int n = n0 + ty + r * 8;
        bt[n * IN_F + (k0 + tx)] = (_Float16)tile[tx][ty + r * 8]; // coalesced write
    }
}

// ---------------------------------------------------------------------------
// WMMA fragment loaders (CDNA5 ISA 7.12.2 wave32 layouts).
//   A (16x32 f16): lane m=lane%16; halves 0-7 = K[g..g+7], 8-15 = K[g+16..g+23],
//                  g = 8*(lane>=16)   -> two 16B loads, 16 halves apart
//   B (32x16 f16): lane n=lane%16; halves 0-7 = K[g..g+7], 8-15 = K[g+16..g+23],
//                  g = 8*(lane>=16)   -> two 16B loads, 16 halves apart
// ---------------------------------------------------------------------------
__device__ __forceinline__ v16h load_frag_a(const _Float16* p) {
    v16h a;
    *(v8h*)&a         = *(const v8h*)(p);
    *(((v8h*)&a) + 1) = *(const v8h*)(p + 16);
    return a;
}
__device__ __forceinline__ v16h load_frag_b(const _Float16* p) {
    v16h b;
    *(v8h*)&b         = *(const v8h*)(p);
    *(((v8h*)&b) + 1) = *(const v8h*)(p + 16);
    return b;
}
#define WMMA_F16(A, Bf, C) \
    __builtin_amdgcn_wmma_f32_16x16x32_f16(false, (A), false, (Bf), (short)0, (C), false, false)

// ---------------------------------------------------------------------------
// Kernel 3: GEMM  M[256,3200] f32 = A16[256,1024] x Bt[3200,1024]^T
//   One wave per 16(M) x 64(N) strip; A fragment shared by 4 WMMAs per K-chunk.
//   Register double-buffering: loads for K-chunk t+1 are issued before the
//   WMMAs of chunk t, so the s_wait before each WMMA group covers loads that
//   already aged a full phase (removes the s_wait_loadcnt 0 -> wmma lockstep).
//   16 M-tiles * 50 N-strips = 800 waves = 100 blocks of 8 waves.
// ---------------------------------------------------------------------------
__global__ void gemm_wmma_kernel(const _Float16* __restrict__ A,
                                 const _Float16* __restrict__ Bt,
                                 float* __restrict__ M)
{
    const int lane  = threadIdx.x & 31;
    const int wid   = blockIdx.x * 8 + (threadIdx.x >> 5);  // 0..799
    const int mtile = wid / 50;
    const int m0    = mtile * 16;
    const int n0    = (wid % 50) * 64;

    const int lmod  = lane & 15;
    const int lhalf = lane >> 4;          // 0 or 1

    // Per-lane base pointers (fragment-layout offsets folded in).
    const _Float16* ap = A  + (m0 + lmod) * IN_F + lhalf * 8;
    const _Float16* bp = Bt + (n0 + lmod) * IN_F + lhalf * 8;
    // N-subtile row strides are compile-time constants -> immediate offsets.
    const int BS = 16 * IN_F;             // 16384 halves = 32768 B

    v8f acc0 = {}, acc1 = {}, acc2 = {}, acc3 = {};

    // ---- pipeline prologue: buffer 0 @ k=0 ----
    v16h a0  = load_frag_a(ap);
    v16h p00 = load_frag_b(bp + 0 * BS);
    v16h p10 = load_frag_b(bp + 1 * BS);
    v16h p20 = load_frag_b(bp + 2 * BS);
    v16h p30 = load_frag_b(bp + 3 * BS);
    v16h a1, p01, p11, p21, p31;

    // 15 iterations: k0 = 0, 64, ..., 896
    #pragma unroll 1
    for (int k0 = 0; k0 < 960; k0 += 64) {
        // prefetch buffer 1 @ k0+32
        a1  = load_frag_a(ap + k0 + 32);
        p01 = load_frag_b(bp + k0 + 32 + 0 * BS);
        p11 = load_frag_b(bp + k0 + 32 + 1 * BS);
        p21 = load_frag_b(bp + k0 + 32 + 2 * BS);
        p31 = load_frag_b(bp + k0 + 32 + 3 * BS);
        // consume buffer 0 @ k0
        acc0 = WMMA_F16(a0, p00, acc0);
        acc1 = WMMA_F16(a0, p10, acc1);
        acc2 = WMMA_F16(a0, p20, acc2);
        acc3 = WMMA_F16(a0, p30, acc3);
        // prefetch buffer 0 @ k0+64
        a0  = load_frag_a(ap + k0 + 64);
        p00 = load_frag_b(bp + k0 + 64 + 0 * BS);
        p10 = load_frag_b(bp + k0 + 64 + 1 * BS);
        p20 = load_frag_b(bp + k0 + 64 + 2 * BS);
        p30 = load_frag_b(bp + k0 + 64 + 3 * BS);
        // consume buffer 1 @ k0+32
        acc0 = WMMA_F16(a1, p01, acc0);
        acc1 = WMMA_F16(a1, p11, acc1);
        acc2 = WMMA_F16(a1, p21, acc2);
        acc3 = WMMA_F16(a1, p31, acc3);
    }
    // ---- epilogue: buffer 0 holds k=960; fetch and consume k=992 ----
    a1  = load_frag_a(ap + 992);
    p01 = load_frag_b(bp + 992 + 0 * BS);
    p11 = load_frag_b(bp + 992 + 1 * BS);
    p21 = load_frag_b(bp + 992 + 2 * BS);
    p31 = load_frag_b(bp + 992 + 3 * BS);
    acc0 = WMMA_F16(a0, p00, acc0);
    acc1 = WMMA_F16(a0, p10, acc1);
    acc2 = WMMA_F16(a0, p20, acc2);
    acc3 = WMMA_F16(a0, p30, acc3);
    acc0 = WMMA_F16(a1, p01, acc0);
    acc1 = WMMA_F16(a1, p11, acc1);
    acc2 = WMMA_F16(a1, p21, acc2);
    acc3 = WMMA_F16(a1, p31, acc3);

    // Store: lane covers columns n0 + {0,16,32,48} + lmod,
    // rows m0 + 8*lhalf + r (r = VGPR index in the C fragment).
    float* o = M + (m0 + lhalf * 8) * NCOLS + n0 + lmod;
    #pragma unroll
    for (int r = 0; r < 8; ++r) {
        o[r * NCOLS +  0] = acc0[r];
        o[r * NCOLS + 16] = acc1[r];
        o[r * NCOLS + 32] = acc2[r];
        o[r * NCOLS + 48] = acc3[r];
    }
}

// ---------------------------------------------------------------------------
// Kernel 4: all-pairs L1 + exp reduction, one block per out_feature o.
//   LDS holds M[:, o*50:(o+1)*50] (51.2 KB; row stride 50 dwords ->
//   gcd(50,64)=2 -> 32 distinct banks across a wave32: conflict-free).
//   Each thread keeps its own 50-float row in VGPRs; j-loop reads are
//   wave-uniform LDS broadcasts.
// ---------------------------------------------------------------------------
__global__ void pairwise_kernel(const float* __restrict__ M,
                                float* __restrict__ out)
{
    __shared__ float s[B_SZ * KD];          // 51,200 B
    const int o = blockIdx.x;               // 0..63
    const int i = threadIdx.x;              // 0..255

    const float* mrow = M + i * NCOLS + o * KD;
    float r[KD];
    #pragma unroll
    for (int k = 0; k < KD; ++k) {
        float v = mrow[k];
        r[k] = v;
        s[i * KD + k] = v;
    }
    __syncthreads();

    float acc = 0.0f;
    for (int j = 0; j < B_SZ; ++j) {
        const float* sj = s + j * KD;
        float l1a = 0.0f, l1b = 0.0f;       // split accumulators for ILP
        #pragma unroll
        for (int k = 0; k < KD; k += 2) {
            l1a += fabsf(r[k]     - sj[k]);
            l1b += fabsf(r[k + 1] - sj[k + 1]);
        }
        // exp(-l1) = exp2(-l1 * log2(e))  -> v_exp_f32
        acc += __builtin_exp2f((l1a + l1b) * -1.442695040888963f);
    }
    // includes j==i term exp(0)=1; reference subtracts it.
    out[i * OUT_W + IN_F + o] = acc - 1.0f;
}

// ---------------------------------------------------------------------------
// Launch
// ---------------------------------------------------------------------------
extern "C" void kernel_launch(void* const* d_in, const int* in_sizes, int n_in,
                              void* d_out, int out_size, void* d_ws, size_t ws_size,
                              hipStream_t stream)
{
    const float* x = (const float*)d_in[0];   // 256*1024
    const float* T = (const float*)d_in[1];   // 1024*3200
    float* out = (float*)d_out;               // 256*1088

    // Workspace layout (bytes, 256-B aligned):
    //   A16 : 256*1024*2  =   524,288
    //   Bt  : 3200*1024*2 = 6,553,600
    //   M   : 256*3200*4  = 3,276,800      total ~10.4 MB
    char* ws = (char*)d_ws;
    _Float16* a16 = (_Float16*)(ws);
    _Float16* bt  = (_Float16*)(ws + 524288);
    float*    M   = (float*)   (ws + 524288 + 6553600);

    // 1) x -> f16, and concat-copy x into out[:, :1024]
    convert_x_kernel<<<(B_SZ * IN_F + 255) / 256, 256, 0, stream>>>(x, a16, out);

    // 2) T -> Bt (transpose + f16 convert); grid (3200/32, 1024/32)
    transpose_T_kernel<<<dim3(NCOLS / 32, IN_F / 32), dim3(32, 8), 0, stream>>>(T, bt);

    // 3) WMMA GEMM: 800 waves = 100 blocks * 8 waves
    gemm_wmma_kernel<<<100, 256, 0, stream>>>(a16, bt, M);

    // 4) pairwise L1 + exp: one block per out_feature
    pairwise_kernel<<<OUT_F, B_SZ, 0, stream>>>(M, out);
}
